// DeformableTransformerEncoderLayer_4861902979347
// MI455X (gfx1250) — matrix operand order repaired and checked by reading, not compiled, not run on hardware
//
#include <hip/hip_runtime.h>

// ---------------- problem constants ----------------
#define D_MODEL   256
#define D_FFN     1024
#define N_LEVELS  4
#define N_HEADS   8
#define N_POINTS  4
#define HEAD_DIM  32
#define LEN_IN    15360
#define BATCH     2
#define NM        (BATCH * LEN_IN)     // 30720 rows
#define LN_EPS    1e-5f

// LDS B-tile geometry: 128 cols x 32 k bf16, column stride padded to 80B
#define COL_BYTES   80                 // 64 data + 16 pad (bank-conflict free)
#define BUF_BYTES   (128 * COL_BYTES)  // 10240 B per buffer
#define BSM_ELEMS   (2 * BUF_BYTES / 2)

// ---------------- vector types for WMMA ----------------
typedef __bf16 v8bf  __attribute__((ext_vector_type(8)));
typedef __bf16 v16bf __attribute__((ext_vector_type(16)));
typedef float  v8f   __attribute__((ext_vector_type(8)));

// =====================================================================
// Prep: q = bf16(src+pos), s = bf16(src)
// =====================================================================
__global__ void dte_prep_act(const float* __restrict__ src,
                             const float* __restrict__ pos,
                             __bf16* __restrict__ q16,
                             __bf16* __restrict__ s16,
                             int n) {
    int i = blockIdx.x * blockDim.x + threadIdx.x;
    if (i >= n) return;
    float s = src[i];
    q16[i] = (__bf16)(s + pos[i]);
    s16[i] = (__bf16)s;
}

// =====================================================================
// Weight prep: Wt[n][k] = bf16(W[k][n])   (W is K x N row-major)
// =====================================================================
__global__ void dte_wtrans(const float* __restrict__ W,
                           __bf16* __restrict__ Wt,
                           int K, int N) {
    int i = blockIdx.x * blockDim.x + threadIdx.x;
    if (i >= K * N) return;
    int k = i / N;
    int n = i - k * N;
    Wt[(size_t)n * K + k] = (__bf16)W[i];
}

// =====================================================================
// bf16 WMMA GEMM:  C[M,N] = A[M,K] * Bt[N,K]^T + bias, optional relu,
// f32 and/or bf16 outputs.
//   Block = 256 thr = 8 waves, block tile 128(M) x 128(N).
//   Waves arranged 4(M) x 2(N); each wave computes 32x64 (8 WMMA accs).
//   B tile staged in LDS with global_load_async_to_lds_b128 (ASYNCcnt),
//   double-buffered; consumed with inline-asm ds_load_b128 so the
//   asm-written LDS cannot be folded away by the compiler.
// Requires: M%128==0, N%128==0, K%32==0 (true for all launches here).
// =====================================================================
__global__ __launch_bounds__(256)
void dte_gemm_bf16(const __bf16* __restrict__ A,
                   const __bf16* __restrict__ Bt,
                   const float*  __restrict__ bias,
                   float*  __restrict__ Cf,
                   __bf16* __restrict__ Cb,
                   int M, int N, int K, int relu) {
    __shared__ __attribute__((aligned(16))) __bf16 bsm[BSM_ELEMS];
    // low 32 bits of a generic shared pointer == LDS byte offset;
    // also keeps the LDS allocation alive (all access is via inline asm).
    const unsigned bsmBase = (unsigned)(uintptr_t)bsm;

    const int t    = threadIdx.x;
    const int wid  = t >> 5;
    const int lane = t & 31;
    const int half = lane >> 4;        // 0 or 1
    const int l15  = lane & 15;
    const int mw   = wid >> 1;         // 0..3 : 32-row group
    const int nw   = wid & 1;          // 0..1 : 64-col group

    const int rowBase = blockIdx.x * 128 + mw * 32;
    const int colBase = blockIdx.y * 128;

    // --- async B-tile producer: each thread copies one 32-byte slice ---
    const int bc = t >> 1;             // column 0..127 within tile
    const int bh = t & 1;              // which 16-k half
    const unsigned prodOff = bsmBase + (unsigned)(bc * COL_BYTES + bh * 32);

    // A fragment rows for this lane (two 16-row groups)
    const __bf16* __restrict__ Arow0 = A + (size_t)(rowBase + l15) * K;
    const __bf16* __restrict__ Arow1 = Arow0 + (size_t)16 * K;

    v8f acc[2][4];
#pragma unroll
    for (int mi = 0; mi < 2; ++mi)
#pragma unroll
        for (int j = 0; j < 4; ++j)
#pragma unroll
            for (int e = 0; e < 8; ++e) acc[mi][j][e] = 0.0f;

    const int nsteps = K >> 5;

    // consumer LDS byte addresses for the 4 B fragments of this wave
    unsigned bAddr[4];
#pragma unroll
    for (int j = 0; j < 4; ++j) {
        const int nLocal = nw * 64 + j * 16 + l15;
        bAddr[j] = bsmBase + (unsigned)(nLocal * COL_BYTES + 32 * half);
    }

    // prologue: issue async copy of B tile for k-step 0 into buffer 0
    {
        const __bf16* g = Bt + (size_t)(colBase + bc) * K + 16 * bh;
        asm volatile("global_load_async_to_lds_b128 %0, %1, off"
                     :: "v"(prodOff), "v"(g) : "memory");
        asm volatile("global_load_async_to_lds_b128 %0, %1, off offset:16"
                     :: "v"(prodOff), "v"(g) : "memory");
    }

    for (int step = 0; step < nsteps; ++step) {
        const unsigned curOff = (unsigned)((step & 1) * BUF_BYTES);
        const int k0 = step << 5;

        // my async copies done -> then all waves' copies done
        asm volatile("s_wait_asynccnt 0" ::: "memory");
        __syncthreads();

        // issue next B tile into the other buffer (overlaps the WMMAs below)
        if (step + 1 < nsteps) {
            const __bf16* g = Bt + (size_t)(colBase + bc) * K + (k0 + 32) + 16 * bh;
            const unsigned l = prodOff + (unsigned)BUF_BYTES - 2u * curOff
                             + curOff;              // prodOff + other-buffer offset
            const unsigned lo = prodOff + ((unsigned)BUF_BYTES - curOff);
            (void)l;
            asm volatile("global_load_async_to_lds_b128 %0, %1, off"
                         :: "v"(lo), "v"(g) : "memory");
            asm volatile("global_load_async_to_lds_b128 %0, %1, off offset:16"
                         :: "v"(lo), "v"(g) : "memory");
        }

        // ---- A fragments (direct from global; wave-private rows)
        const int ka = k0 + 16 * half;
        if (step + 1 < nsteps) {
            __builtin_prefetch(Arow0 + ka + 32, 0, 3);
            __builtin_prefetch(Arow1 + ka + 32, 0, 3);
        }
        v16bf afrag[2];
        {
            v8bf lo0 = *(const v8bf*)(Arow0 + ka);
            v8bf hi0 = *(const v8bf*)(Arow0 + ka + 8);
            v8bf lo1 = *(const v8bf*)(Arow1 + ka);
            v8bf hi1 = *(const v8bf*)(Arow1 + ka + 8);
#pragma unroll
            for (int e = 0; e < 8; ++e) {
                afrag[0][e] = lo0[e]; afrag[0][e + 8] = hi0[e];
                afrag[1][e] = lo1[e]; afrag[1][e + 8] = hi1[e];
            }
        }

        // ---- B fragments from LDS via inline-asm ds_load_b128.
        // The trailing s_wait_dscnt lives in the SAME asm statement, so the
        // WMMAs (which consume the asm outputs) cannot start before the wait.
        v8bf bl0, bh0, bl1, bh1, bl2, bh2, bl3, bh3;
        asm volatile(
            "ds_load_b128 %0, %8\n\t"
            "ds_load_b128 %1, %8 offset:16\n\t"
            "ds_load_b128 %2, %9\n\t"
            "ds_load_b128 %3, %9 offset:16\n\t"
            "ds_load_b128 %4, %10\n\t"
            "ds_load_b128 %5, %10 offset:16\n\t"
            "ds_load_b128 %6, %11\n\t"
            "ds_load_b128 %7, %11 offset:16\n\t"
            "s_wait_dscnt 0x0"
            : "=&v"(bl0), "=&v"(bh0), "=&v"(bl1), "=&v"(bh1),
              "=&v"(bl2), "=&v"(bh2), "=&v"(bl3), "=&v"(bh3)
            : "v"(bAddr[0] + curOff), "v"(bAddr[1] + curOff),
              "v"(bAddr[2] + curOff), "v"(bAddr[3] + curOff)
            : "memory");

        v16bf bfrag[4];
#pragma unroll
        for (int e = 0; e < 8; ++e) {
            bfrag[0][e] = bl0[e]; bfrag[0][e + 8] = bh0[e];
            bfrag[1][e] = bl1[e]; bfrag[1][e + 8] = bh1[e];
            bfrag[2][e] = bl2[e]; bfrag[2][e + 8] = bh2[e];
            bfrag[3][e] = bl3[e]; bfrag[3][e + 8] = bh3[e];
        }

#pragma unroll
        for (int mi = 0; mi < 2; ++mi)
#pragma unroll
            for (int j = 0; j < 4; ++j)
                acc[mi][j] = __builtin_amdgcn_wmma_f32_16x16x32_bf16(
                    false, afrag[mi], false, bfrag[j], (short)0, acc[mi][j],
                    false, false);

        // all waves done reading buf[cur] before it can be refilled
        __syncthreads();
    }

    // ---- epilogue: D layout -> vgpr v holds M = v + 8*half, lane -> N
#pragma unroll
    for (int mi = 0; mi < 2; ++mi) {
#pragma unroll
        for (int j = 0; j < 4; ++j) {
            const int col = colBase + nw * 64 + j * 16 + l15;
            const float bs = bias ? bias[col] : 0.0f;
#pragma unroll
            for (int v = 0; v < 8; ++v) {
                const int row = rowBase + mi * 16 + v + 8 * half;
                float x = acc[mi][j][v] + bs;
                if (relu) x = fmaxf(x, 0.0f);
                const size_t o = (size_t)row * N + col;
                if (Cf) Cf[o] = x;
                if (Cb) Cb[o] = (__bf16)x;
            }
        }
    }
}

// =====================================================================
// Deformable sampling. One wave32 per (row, head); lane = channel d.
// =====================================================================
__global__ __launch_bounds__(256)
void dte_sample(const float* __restrict__ value,   // [NM,256] f32
                const float* __restrict__ offv,    // [NM,128]
                const float* __restrict__ logit,   // [NM,128]
                const float* __restrict__ refp,    // [NM,4]
                __bf16* __restrict__ attn16) {     // [NM,256]
    const int lens[N_LEVELS]   = {8192, 4096, 2048, 1024};
    const int starts[N_LEVELS] = {0, 8192, 12288, 14336};

    const int wid  = threadIdx.x >> 5;
    const int lane = threadIdx.x & 31;
    const int gid  = blockIdx.x * 8 + wid;   // over NM * N_HEADS waves
    const int r    = gid >> 3;
    const int h    = gid & 7;
    const int nb   = r / LEN_IN;             // batch index

    const float* __restrict__ lg = logit + (size_t)r * 128 + h * 16;
    const float* __restrict__ of = offv  + (size_t)r * 128 + h * 16;
    const float* __restrict__ rp = refp  + (size_t)r * 4;

    float e[16];
    float mx = -3.0e38f;
#pragma unroll
    for (int i = 0; i < 16; ++i) { e[i] = lg[i]; mx = fmaxf(mx, e[i]); }
    float ssum = 0.0f;
#pragma unroll
    for (int i = 0; i < 16; ++i) { e[i] = __expf(e[i] - mx); ssum += e[i]; }
    const float inv = 1.0f / ssum;

    float out = 0.0f;
#pragma unroll
    for (int l = 0; l < N_LEVELS; ++l) {
        const int   T  = lens[l];
        const float Tf = (float)T;
        const float rf = rp[l];
        const float* __restrict__ vb =
            value + ((size_t)nb * LEN_IN + starts[l]) * D_MODEL + h * HEAD_DIM + lane;
#pragma unroll
        for (int p = 0; p < N_POINTS; ++p) {
            const int   i_ = l * N_POINTS + p;
            const float w  = e[i_] * inv;
            // grid-sample coord math collapses to: ix = ref*T + off - 0.5
            const float ix = rf * Tf + of[i_] - 0.5f;
            const float x0 = floorf(ix);
            const float fr = ix - x0;
            const int   i0 = (int)x0;
            const int   i1 = i0 + 1;
            float v0 = (i0 >= 0 && i0 < T) ? vb[(size_t)i0 * D_MODEL] : 0.0f;
            float v1 = (i1 >= 0 && i1 < T) ? vb[(size_t)i1 * D_MODEL] : 0.0f;
            out += w * (v0 * (1.0f - fr) + v1 * fr);
        }
    }
    attn16[(size_t)r * D_MODEL + h * HEAD_DIM + lane] = (__bf16)out;
}

// =====================================================================
// x = LN(a+b)*g + beta (+ add_after);  one 256-thread block per row.
// =====================================================================
__global__ __launch_bounds__(256)
void dte_add_ln(const float* __restrict__ a,
                const float* __restrict__ b,
                const float* __restrict__ g,
                const float* __restrict__ beta,
                const float* __restrict__ add_after,
                float*  __restrict__ outf,
                __bf16* __restrict__ outb) {
    const int row  = blockIdx.x;
    const int c    = threadIdx.x;           // 0..255 == D_MODEL
    const int wid  = c >> 5;
    const int lane = c & 31;
    const size_t o = (size_t)row * D_MODEL + c;

    float v = a[o] + b[o];

    float s1 = v, s2 = v * v;
#pragma unroll
    for (int off = 16; off > 0; off >>= 1) {
        s1 += __shfl_xor(s1, off, 32);
        s2 += __shfl_xor(s2, off, 32);
    }
    __shared__ float sh1[8], sh2[8];
    if (lane == 0) { sh1[wid] = s1; sh2[wid] = s2; }
    __syncthreads();
    if (c == 0) {
        float t1 = 0.0f, t2 = 0.0f;
#pragma unroll
        for (int w = 0; w < 8; ++w) { t1 += sh1[w]; t2 += sh2[w]; }
        sh1[0] = t1; sh2[0] = t2;
    }
    __syncthreads();
    const float mu  = sh1[0] * (1.0f / D_MODEL);
    const float var = sh2[0] * (1.0f / D_MODEL) - mu * mu;
    float y = (v - mu) * rsqrtf(var + LN_EPS) * g[c] + beta[c];
    if (add_after) y += add_after[o];
    if (outf) outf[o] = y;
    if (outb) outb[o] = (__bf16)y;
}

// =====================================================================
// Launch
// =====================================================================
extern "C" void kernel_launch(void* const* d_in, const int* in_sizes, int n_in,
                              void* d_out, int out_size, void* d_ws, size_t ws_size,
                              hipStream_t stream) {
    const float* src    = (const float*)d_in[0];
    const float* pos    = (const float*)d_in[1];
    const float* t_emb  = (const float*)d_in[2];
    const float* refp   = (const float*)d_in[3];
    const float* w_off  = (const float*)d_in[4];
    const float* b_off  = (const float*)d_in[5];
    const float* w_attn = (const float*)d_in[6];
    const float* b_attn = (const float*)d_in[7];
    const float* w_val  = (const float*)d_in[8];
    const float* b_val  = (const float*)d_in[9];
    const float* w_out  = (const float*)d_in[10];
    const float* b_out  = (const float*)d_in[11];
    const float* ln1_g  = (const float*)d_in[12];
    const float* ln1_b  = (const float*)d_in[13];
    const float* w1     = (const float*)d_in[14];
    const float* b1     = (const float*)d_in[15];
    const float* w2     = (const float*)d_in[16];
    const float* b2     = (const float*)d_in[17];
    const float* ln2_g  = (const float*)d_in[18];
    const float* ln2_b  = (const float*)d_in[19];
    float* out = (float*)d_out;

    // ---------------- workspace carve-up ----------------
    char* ws = (char*)d_ws;
    size_t off = 0;
    auto carve = [&](size_t bytes) { char* p = ws + off; off += (bytes + 255) & ~(size_t)255; return p; };

    __bf16* q16   = (__bf16*)carve((size_t)NM * D_MODEL * 2);
    __bf16* s16   = (__bf16*)carve((size_t)NM * D_MODEL * 2);
    __bf16* wtv   = (__bf16*)carve((size_t)D_MODEL * D_MODEL * 2);
    __bf16* wto   = (__bf16*)carve((size_t)128 * D_MODEL * 2);
    __bf16* wta   = (__bf16*)carve((size_t)128 * D_MODEL * 2);
    __bf16* wtO   = (__bf16*)carve((size_t)D_MODEL * D_MODEL * 2);
    __bf16* wt1   = (__bf16*)carve((size_t)D_FFN * D_MODEL * 2);
    __bf16* wt2   = (__bf16*)carve((size_t)D_MODEL * D_FFN * 2);
    // value/off/logit region -- reused as FFN hidden (h16) after sampling
    char*  reuse0 = carve((size_t)NM * D_MODEL * 4      // value f32
                        + (size_t)NM * 128 * 4          // off
                        + (size_t)NM * 128 * 4);        // logits
    float*  value = (float*)reuse0;
    float*  offv  = (float*)(reuse0 + (size_t)NM * D_MODEL * 4);
    float*  logit = (float*)(reuse0 + (size_t)NM * D_MODEL * 4 + (size_t)NM * 128 * 4);
    __bf16* h16   = (__bf16*)reuse0;                    // NM*1024 bf16 = same bytes
    __bf16* a16   = (__bf16*)carve((size_t)NM * D_MODEL * 2);   // attn out (bf16)
    // src2 region reused for ff
    float*  src2  = (float*)carve((size_t)NM * D_MODEL * 4);
    float*  ff    = src2;
    float*  x32   = (float*)carve((size_t)NM * D_MODEL * 4);
    __bf16* x16   = (__bf16*)carve((size_t)NM * D_MODEL * 2);
    (void)ws_size; (void)in_sizes; (void)n_in; (void)out_size;

    const int nElem = NM * D_MODEL;

    // 1) activations -> bf16
    dte_prep_act<<<(nElem + 255) / 256, 256, 0, stream>>>(src, pos, q16, s16, nElem);

    // 2) weights -> transposed bf16
    dte_wtrans<<<(D_MODEL * D_MODEL + 255) / 256, 256, 0, stream>>>(w_val,  wtv, D_MODEL, D_MODEL);
    dte_wtrans<<<(D_MODEL * 128     + 255) / 256, 256, 0, stream>>>(w_off,  wto, D_MODEL, 128);
    dte_wtrans<<<(D_MODEL * 128     + 255) / 256, 256, 0, stream>>>(w_attn, wta, D_MODEL, 128);
    dte_wtrans<<<(D_MODEL * D_MODEL + 255) / 256, 256, 0, stream>>>(w_out,  wtO, D_MODEL, D_MODEL);
    dte_wtrans<<<(D_MODEL * D_FFN   + 255) / 256, 256, 0, stream>>>(w1,     wt1, D_MODEL, D_FFN);
    dte_wtrans<<<(D_FFN * D_MODEL   + 255) / 256, 256, 0, stream>>>(w2,     wt2, D_FFN, D_MODEL);

    const dim3 blk(256);
    // 3) value = src @ w_val + b_val           (30720x256x256)
    dte_gemm_bf16<<<dim3(NM / 128, D_MODEL / 128), blk, 0, stream>>>(
        s16, wtv, b_val, value, nullptr, NM, D_MODEL, D_MODEL, 0);
    // 4) off = q @ w_off + b_off               (30720x128x256)
    dte_gemm_bf16<<<dim3(NM / 128, 1), blk, 0, stream>>>(
        q16, wto, b_off, offv, nullptr, NM, 128, D_MODEL, 0);
    // 5) attn logits = q @ w_attn + b_attn     (30720x128x256)
    dte_gemm_bf16<<<dim3(NM / 128, 1), blk, 0, stream>>>(
        q16, wta, b_attn, logit, nullptr, NM, 128, D_MODEL, 0);

    // 6) deformable sampling -> a16
    dte_sample<<<NM, 256, 0, stream>>>(value, offv, logit, refp, a16);

    // 7) src2 = a16 @ w_out + b_out            (30720x256x256)
    dte_gemm_bf16<<<dim3(NM / 128, D_MODEL / 128), blk, 0, stream>>>(
        a16, wtO, b_out, src2, nullptr, NM, D_MODEL, D_MODEL, 0);

    // 8) x = LN(src + src2) + time_embed  -> x32, x16
    dte_add_ln<<<NM, 256, 0, stream>>>(src, src2, ln1_g, ln1_b, t_emb, x32, x16);

    // 9) h = relu(x @ w1 + b1)  -> bf16        (30720x1024x256)
    dte_gemm_bf16<<<dim3(NM / 128, D_FFN / 128), blk, 0, stream>>>(
        x16, wt1, b1, nullptr, h16, NM, D_FFN, D_MODEL, 1);
    // 10) ff = h @ w2 + b2                     (30720x256x1024)
    dte_gemm_bf16<<<dim3(NM / 128, D_MODEL / 128), blk, 0, stream>>>(
        h16, wt2, b2, ff, nullptr, NM, D_MODEL, D_FFN, 0);

    // 11) out = LN(x + ff)
    dte_add_ln<<<NM, 256, 0, stream>>>(x32, ff, ln2_g, ln2_b, nullptr, out, nullptr);
}
